// EERNNModel_51857435132233
// MI455X (gfx1250) — matrix operand order repaired and checked by reading, not compiled, not run on hardware
//
#include <hip/hip_runtime.h>
#include <math.h>

// ---------------- model dims (compile-time) ----------------
#define EMB   128
#define EXC_D 256
#define HID   256
#define LSEQ  64
#define NROWS 200000
#define TOPK  10
#define H2    128      // per-direction hidden
#define G3    384      // 3*H2 gates

typedef __attribute__((ext_vector_type(16))) _Float16 v16h;
typedef __attribute__((ext_vector_type(8)))  float    v8f;

// =====================================================================
// Kernel 1: batched input projection Gi[dir][t][g] = x @ Wih.T
//   x[t][k] = emb[exec_ids[t]][k]   (gathered on the fly)
//   one wave per 16x16 output tile; K=128 -> 4 chained WMMAs (f16 in, f32 acc)
//   Fragment layouts per CDNA5 ISA 7.12.2 (wave32):
//     A(16x32): m = lane&15 ; K = kb + e + (e>=8?8:0) + (lane>=16?8:0)
//     B(32x16): n = lane&15 ; K = kb + e + (e>=8?8:0) + (lane>=16?8:0)
//     C/D(16x16 f32): m = v + (lane>=16?8:0), n = lane&15
// =====================================================================
__global__ __launch_bounds__(32)
void eernn_proj(const int* __restrict__ exec_ids,
                const float* __restrict__ emb,
                const float* __restrict__ wf_Wih,
                const float* __restrict__ wb_Wih,
                float* __restrict__ gi /* [2][LSEQ][G3] */) {
  const int wave = blockIdx.x;          // 0..191
  const int dir  = wave / 96;
  const int tile = wave % 96;
  const int mt   = tile / 24;           // time tile (0..3)
  const int nt   = tile % 24;           // gate tile (0..23)
  const float* __restrict__ Wih = dir ? wb_Wih : wf_Wih;

  const int l    = threadIdx.x;         // 0..31
  const int half = l >> 4;              // 0 or 1
  const int mrow = mt * 16 + (l & 15);  // time index
  const int ncol = nt * 16 + (l & 15);  // gate index

  const int wid = exec_ids[mrow];
  const float* __restrict__ xrow = emb + (size_t)wid * EMB;   // A row (m fixed per lane)
  const float* __restrict__ wrow = Wih + (size_t)ncol * EMB;  // B col (Wih[n][k])

  v8f c = {};
  #pragma unroll
  for (int kb = 0; kb < EMB; kb += 32) {
    v16h a, b;
    #pragma unroll
    for (int e = 0; e < 16; ++e) {
      const int ka  = kb + e + ((e >= 8) ? 8 : 0) + half * 8;
      const int kbn = kb + e + ((e >= 8) ? 8 : 0) + half * 8;
      a[e] = (_Float16)xrow[ka];
      b[e] = (_Float16)wrow[kbn];
    }
    c = __builtin_amdgcn_wmma_f32_16x16x32_f16(
          /*neg_a=*/false, a, /*neg_b=*/false, b,
          /*c_mod=*/(short)0, c, /*reuse_a=*/false, /*reuse_b=*/false);
  }
  #pragma unroll
  for (int v = 0; v < 8; ++v) {
    const int m = mt * 16 + v + half * 8;
    const int n = nt * 16 + (l & 15);
    gi[((size_t)dir * LSEQ + m) * G3 + n] = c[v];
  }
}

// =====================================================================
// Kernel 2: sequential biGRU recurrence + fused max-pool over time.
// One workgroup, 768 threads: thread = (dir, gate j). Gi precomputed.
// Backward direction consumes Gi[1][63-t] at step t (reversed input).
// =====================================================================
__global__ __launch_bounds__(768)
void eernn_recur(const float* __restrict__ gi,
                 const float* __restrict__ wf_Whh, const float* __restrict__ wf_bih,
                 const float* __restrict__ wf_bhh,
                 const float* __restrict__ wb_Whh, const float* __restrict__ wb_bih,
                 const float* __restrict__ wb_bhh,
                 float* __restrict__ exc_ws,        // [EXC_D] workspace
                 float* __restrict__ out_exc_row) { // excs_new row N
  __shared__ float h[2][H2];
  __shared__ float gh[2][G3];
  const int tid = threadIdx.x;
  const int dir = tid / G3;
  const int j   = tid % G3;
  const float* __restrict__ Whh = dir ? wb_Whh : wf_Whh;
  const float* __restrict__ bih = dir ? wb_bih : wf_bih;
  const float* __restrict__ bhh = dir ? wb_bhh : wf_bhh;
  const float* __restrict__ wrow = Whh + (size_t)j * H2;
  __builtin_prefetch(wrow, 0, 3);  // gfx1250 global_prefetch_b8 (L2-resident reuse x64)

  if (j < H2) h[dir][j] = 0.0f;
  float ymax = -3.0e38f;
  __syncthreads();

  for (int t = 0; t < LSEQ; ++t) {
    const int ti = dir ? (LSEQ - 1 - t) : t;
    // gh[j] = h . Whh[j] + bhh[j]
    float acc = bhh[j];
    #pragma unroll 8
    for (int k = 0; k < H2; ++k) acc += h[dir][k] * wrow[k];
    gh[dir][j] = acc;
    __syncthreads();

    float hnew = 0.0f;
    if (j < H2) {
      const float* __restrict__ g = gi + ((size_t)dir * LSEQ + ti) * G3;
      const float ir = g[j]          + bih[j];
      const float iz = g[H2 + j]     + bih[H2 + j];
      const float in = g[2 * H2 + j] + bih[2 * H2 + j];
      const float hr = gh[dir][j];
      const float hz = gh[dir][H2 + j];
      const float hn = gh[dir][2 * H2 + j];
      const float r = 1.0f / (1.0f + expf(-(ir + hr)));
      const float z = 1.0f / (1.0f + expf(-(iz + hz)));
      const float n = tanhf(in + r * hn);
      hnew = (1.0f - z) * n + z * h[dir][j];
    }
    __syncthreads();           // everyone done reading old h / gh
    if (j < H2) {
      h[dir][j] = hnew;
      ymax = fmaxf(ymax, hnew);
    }
    __syncthreads();           // new h visible for next step
  }
  if (j < H2) {
    exc_ws[dir * H2 + j]      = ymax;
    out_exc_row[dir * H2 + j] = ymax;
  }
}

// =====================================================================
// Kernel 3 (bandwidth-dominant, ~820 MB => ~35 us floor at 23.3 TB/s):
// fused  (a) excs_new[i] = excs[i]  +  alpha[i] = excs[i] . exc
//        (b) hs_new[i]   = hs[i]
// One wave (32 lanes) per 256-float row; 2x float4 per lane; wave-reduce dot.
// =====================================================================
__global__ __launch_bounds__(256)
void eernn_gemv_copy(const float* __restrict__ excs, const float* __restrict__ hs,
                     const float* __restrict__ exc_ws,
                     float* __restrict__ alpha,
                     float* __restrict__ out_excs, float* __restrict__ out_hs) {
  const int gwave = (int)((blockIdx.x * blockDim.x + threadIdx.x) >> 5);
  const int lane  = threadIdx.x & 31;
  if (gwave >= 2 * NROWS) return;
  const bool isH = gwave >= NROWS;
  const int  row = isH ? (gwave - NROWS) : gwave;

  const float4* __restrict__ src =
      (const float4*)((isH ? hs : excs) + (size_t)row * 256);
  float4* __restrict__ dst =
      (float4*)((isH ? out_hs : out_excs) + (size_t)row * 256);

  const float4 v0 = src[lane];
  const float4 v1 = src[lane + 32];
  dst[lane]      = v0;
  dst[lane + 32] = v1;

  if (!isH) {
    const float4* __restrict__ e4 = (const float4*)exc_ws;
    const float4 e0 = e4[lane];
    const float4 e1 = e4[lane + 32];
    float d = v0.x * e0.x + v0.y * e0.y + v0.z * e0.z + v0.w * e0.w
            + v1.x * e1.x + v1.y * e1.y + v1.z * e1.z + v1.w * e1.w;
    #pragma unroll
    for (int off = 16; off > 0; off >>= 1) d += __shfl_xor(d, off, 32);
    if (lane == 0) alpha[row] = d;
  }
}

// =====================================================================
// Kernel 4: top-10 of alpha (200000), softmax, attention gather,
// output linear, and the single score-gated GRU step. One workgroup.
// =====================================================================
__global__ __launch_bounds__(256)
void eernn_tail(const float* __restrict__ alpha, const float* __restrict__ exc_ws,
                const float* __restrict__ hs, const float* __restrict__ score,
                const float* __restrict__ s_Wih, const float* __restrict__ s_Whh,
                const float* __restrict__ s_bih, const float* __restrict__ s_bhh,
                const float* __restrict__ out_W, const float* __restrict__ out_b,
                float* __restrict__ pred, float* __restrict__ out_h_row) {
  __shared__ float cv[256 * TOPK];
  __shared__ int   ci[256 * TOPK];
  __shared__ float red[256];
  __shared__ int   redi[256];
  __shared__ float topv[TOPK];
  __shared__ int   topi[TOPK];
  __shared__ float w[TOPK];
  __shared__ float xin[2 * EXC_D];
  __shared__ float hsh[HID];
  const int tid = threadIdx.x;

  // ---- phase 1: thread-local sorted top-10 over strided scan ----
  float lv[TOPK]; int li[TOPK];
  #pragma unroll
  for (int s = 0; s < TOPK; ++s) { lv[s] = -3.0e38f; li[s] = -1; }
  for (int i = tid; i < NROWS; i += 256) {
    const float v = alpha[i];
    if (v > lv[TOPK - 1]) {
      int p = TOPK - 1;
      #pragma unroll
      for (int s = TOPK - 1; s > 0; --s) {
        if (lv[s - 1] < v) { lv[s] = lv[s - 1]; li[s] = li[s - 1]; p = s - 1; }
      }
      lv[p] = v; li[p] = i;
    }
  }
  #pragma unroll
  for (int s = 0; s < TOPK; ++s) { cv[tid * TOPK + s] = lv[s]; ci[tid * TOPK + s] = li[s]; }
  __syncthreads();

  // ---- phase 2: 10 rounds of global max-extraction over 2560 candidates ----
  for (int k = 0; k < TOPK; ++k) {
    float m = -3.0e38f; int marg = 0;
    #pragma unroll
    for (int s = 0; s < TOPK; ++s) {
      const float v = cv[tid * TOPK + s];
      if (v > m) { m = v; marg = tid * TOPK + s; }
    }
    red[tid] = m; redi[tid] = marg;
    __syncthreads();
    for (int off = 128; off > 0; off >>= 1) {
      if (tid < off && red[tid + off] > red[tid]) {
        red[tid] = red[tid + off]; redi[tid] = redi[tid + off];
      }
      __syncthreads();
    }
    if (tid == 0) {
      const int a = redi[0];
      topv[k] = cv[a]; topi[k] = ci[a];
      cv[a] = -3.0e38f;
    }
    __syncthreads();
  }

  // ---- softmax over top-10 (topv sorted descending => topv[0] is max) ----
  if (tid == 0) {
    const float mx = topv[0];
    float ssum = 0.0f;
    for (int k = 0; k < TOPK; ++k) { w[k] = expf(topv[k] - mx); ssum += w[k]; }
    const float inv = 1.0f / ssum;
    for (int k = 0; k < TOPK; ++k) w[k] *= inv;
  }

  // ---- stage xin = [exc*pos, exc*neg], h = hs[N-1] ----
  const float pos = (score[0] >= 0.5f) ? 1.0f : 0.0f;
  const float ej = exc_ws[tid];
  xin[tid]         = ej * pos;
  xin[EXC_D + tid] = ej * (1.0f - pos);
  hsh[tid] = hs[(size_t)(NROWS - 1) * HID + tid];
  __syncthreads();

  // ---- attn_h[tid] = sum_k w[k] * hs[idx[k]][tid] ----
  float a = 0.0f;
  #pragma unroll
  for (int k = 0; k < TOPK; ++k) a += w[k] * hs[(size_t)topi[k] * HID + tid];

  // ---- pred = [exc, attn_h] . out_W + out_b ----
  red[tid] = ej * out_W[tid] + a * out_W[EXC_D + tid];
  __syncthreads();
  for (int off = 128; off > 0; off >>= 1) {
    if (tid < off) red[tid] += red[tid + off];
    __syncthreads();
  }
  if (tid == 0) pred[0] = red[0] + out_b[0];

  // ---- single GRU step: thread tid owns hidden unit tid (3 gate rows) ----
  float ir = s_bih[tid], iz = s_bih[HID + tid], in = s_bih[2 * HID + tid];
  const float* __restrict__ wr = s_Wih + (size_t)tid * (2 * EXC_D);
  const float* __restrict__ wz = s_Wih + (size_t)(HID + tid) * (2 * EXC_D);
  const float* __restrict__ wn = s_Wih + (size_t)(2 * HID + tid) * (2 * EXC_D);
  #pragma unroll 4
  for (int k = 0; k < 2 * EXC_D; ++k) {
    const float x = xin[k];
    ir += x * wr[k]; iz += x * wz[k]; in += x * wn[k];
  }
  float hr = s_bhh[tid], hz = s_bhh[HID + tid], hn = s_bhh[2 * HID + tid];
  const float* __restrict__ ur = s_Whh + (size_t)tid * HID;
  const float* __restrict__ uz = s_Whh + (size_t)(HID + tid) * HID;
  const float* __restrict__ un = s_Whh + (size_t)(2 * HID + tid) * HID;
  #pragma unroll 4
  for (int k = 0; k < HID; ++k) {
    const float hv = hsh[k];
    hr += hv * ur[k]; hz += hv * uz[k]; hn += hv * un[k];
  }
  const float r = 1.0f / (1.0f + expf(-(ir + hr)));
  const float z = 1.0f / (1.0f + expf(-(iz + hz)));
  const float n = tanhf(in + r * hn);
  out_h_row[tid] = (1.0f - z) * n + z * hsh[tid];
}

// =====================================================================
// launcher
// =====================================================================
extern "C" void kernel_launch(void* const* d_in, const int* in_sizes, int n_in,
                              void* d_out, int out_size, void* d_ws, size_t ws_size,
                              hipStream_t stream) {
  (void)in_sizes; (void)n_in; (void)out_size; (void)ws_size;

  const int*   exec_ids = (const int*)  d_in[0];
  const float* score    = (const float*)d_in[1];
  const float* excs     = (const float*)d_in[2];
  const float* hs       = (const float*)d_in[3];
  const float* emb      = (const float*)d_in[4];
  const float* wf_Wih   = (const float*)d_in[5];
  const float* wf_Whh   = (const float*)d_in[6];
  const float* wf_bih   = (const float*)d_in[7];
  const float* wf_bhh   = (const float*)d_in[8];
  const float* wb_Wih   = (const float*)d_in[9];
  const float* wb_Whh   = (const float*)d_in[10];
  const float* wb_bih   = (const float*)d_in[11];
  const float* wb_bhh   = (const float*)d_in[12];
  const float* s_Wih    = (const float*)d_in[13];
  const float* s_Whh    = (const float*)d_in[14];
  const float* s_bih    = (const float*)d_in[15];
  const float* s_bhh    = (const float*)d_in[16];
  const float* out_W    = (const float*)d_in[17];
  const float* out_b    = (const float*)d_in[18];

  // output layout: pred[1] | excs_new[(N+1)*EXC] | hs_new[(N+1)*HID]
  float* out      = (float*)d_out;
  float* pred     = out;
  float* out_excs = out + 1;
  float* out_hs   = out + 1 + (size_t)(NROWS + 1) * EXC_D;

  // workspace layout: gi[2*64*384] | exc[256] | alpha[200000]  (~1.0 MB)
  float* ws    = (float*)d_ws;
  float* gi    = ws;
  float* exc   = ws + 2 * LSEQ * G3;
  float* alpha = exc + EXC_D;

  // 1) WMMA input projection (192 tiles, one wave each)
  eernn_proj<<<192, 32, 0, stream>>>(exec_ids, emb, wf_Wih, wb_Wih, gi);

  // 2) sequential recurrence + max-pool (also writes excs_new row N)
  eernn_recur<<<1, 768, 0, stream>>>(gi, wf_Whh, wf_bih, wf_bhh,
                                     wb_Whh, wb_bih, wb_bhh,
                                     exc, out_excs + (size_t)NROWS * EXC_D);

  // 3) fused GEMV + streaming copies: 400000 rows, 8 waves/block
  eernn_gemv_copy<<<(2 * NROWS) / 8, 256, 0, stream>>>(excs, hs, exc, alpha,
                                                       out_excs, out_hs);

  // 4) top-k + softmax + attention + output head + GRU step (writes hs_new row N)
  eernn_tail<<<1, 256, 0, stream>>>(alpha, exc, hs, score,
                                    s_Wih, s_Whh, s_bih, s_bhh,
                                    out_W, out_b, pred,
                                    out_hs + (size_t)NROWS * HID);
}
